// VSSM_87196426044023
// MI455X (gfx1250) — matrix operand order repaired and checked by reading, not compiled, not run on hardware
//
#include <hip/hip_runtime.h>
#include <hip/hip_bf16.h>
#include <math.h>

typedef __attribute__((ext_vector_type(16))) __bf16 v16bf;
typedef __attribute__((ext_vector_type(8)))  float  v8f;

#define BB   8
#define LQ   4096          // H*W (64x64)
#define DD   384
#define DIN  768
#define KDIR 4
#define NST  16
#define RRK  24
#define CPROJ 56           // R + 2N
#define LDK  40            // padded column stride (bf16 elems) for B tiles in LDS

union Frag  { v16bf v; uint4 q[2]; };
union Acc8  { v8f  v; float4 f[2]; };

__device__ __forceinline__ uint pack2(float a, float b)
{
    union { __bf16 h[2]; uint u; } p;
    p.h[0] = (__bf16)a; p.h[1] = (__bf16)b;
    return p.u;
}

// A fragment (16x32, row-major LDS, lda in elems): lane r = row M,
// elems 0..7 -> K = half*8+i, elems 8..15 -> K = 16+half*8+i. Both 16B chunks.
__device__ __forceinline__ v16bf load_frag_a(const __bf16* base, int lda)
{
    const int lane = threadIdx.x & 31;
    const int half = lane >> 4, r = lane & 15;
    Frag f;
    const __bf16* p = base + r * lda + half * 8;
    f.q[0] = *(const uint4*)(p);
    f.q[1] = *(const uint4*)(p + 16);
    return f.v;
}

// B fragment (32x16) from column-major LDS tile (col stride LDK):
// lane r = col N, elems 0..7 -> K = half*8+i, elems 8..15 -> K = 16+half*8+i.
__device__ __forceinline__ v16bf load_frag_b(const __bf16* base)
{
    const int lane = threadIdx.x & 31;
    const int half = lane >> 4, r = lane & 15;
    Frag f;
    const __bf16* p = base + r * LDK + half * 8;
    f.q[0] = *(const uint4*)(p);
    f.q[1] = *(const uint4*)(p + 16);
    return f.v;
}

// direction-dependent index map (involution): read pos == write pos
__device__ __forceinline__ int pos_map(int k, int l)
{
    int p = l;
    if (k >= 2) p = (LQ - 1) - p;                 // flip
    if (k & 1)  p = ((p & 63) << 6) | (p >> 6);   // H<->W transpose (64x64)
    return p;
}

// -------------------------------------------------------------------------
// one-time: transpose f32 (K x N) -> bf16 (N x K)
// -------------------------------------------------------------------------
__global__ void k_transpose_cvt(const float* __restrict__ src, __bf16* __restrict__ dst,
                                int K, int N)
{
    int idx = blockIdx.x * 256 + threadIdx.x;
    if (idx >= K * N) return;
    int k = idx / N, n = idx - k * N;
    dst[(size_t)n * K + k] = (__bf16)src[idx];
}

// -------------------------------------------------------------------------
// GEMM 1: xz = x(32768x384) @ W_in(384x1536), B pre-transposed bf16 (1536x384).
// block tile 64x64, 8 waves (4 M x 2 N), wave = 16x32, K step 32.
// split epilogue: cols [0,768) -> xm_t (b,d,l) ; cols [768,1536) -> z (b,l,d)
// N-tile never straddles 768 -> uniform branch.
// -------------------------------------------------------------------------
__global__ __launch_bounds__(256) void k_gemm_in(const float* __restrict__ x,
                                                 const __bf16* __restrict__ Wt,
                                                 float* __restrict__ xm_t,
                                                 float* __restrict__ z)
{
    __shared__ __bf16 As[64 * 32];
    __shared__ __bf16 Bt[64 * LDK];
    const int m0 = blockIdx.x * 64;
    const int n0 = blockIdx.y * 64;
    const int tid = threadIdx.x;
    const int lane = tid & 31, wv = tid >> 5;
    const int wm = wv >> 1, wn = wv & 1;
    const int half = lane >> 4, r = lane & 15;
    v8f acc0 = {}; v8f acc1 = {};

    const int arow = tid >> 3, acol = (tid & 7) * 4;          // A: 2 float4 / thread
    const int brow = tid >> 2, bcol = (tid & 3) * 8;          // B: 1 uint4 / thread

    for (int k0 = 0; k0 < 384; k0 += 32) {
        if (k0 + 32 < 384) {
            __builtin_prefetch(&x[(size_t)(m0 + arow) * 384 + k0 + 32 + acol], 0, 1);
            __builtin_prefetch(&Wt[(size_t)(n0 + brow) * 384 + k0 + 32 + bcol], 0, 1);
        }
#pragma unroll
        for (int c = 0; c < 2; ++c) {                          // A tile 64x32 f32->bf16
            int row = arow + c * 32;
            float4 g = *(const float4*)&x[(size_t)(m0 + row) * 384 + k0 + acol];
            uint2 s; s.x = pack2(g.x, g.y); s.y = pack2(g.z, g.w);
            *(uint2*)&As[row * 32 + acol] = s;
        }
        {                                                      // B tile 32x64, col-major
            uint4 g = *(const uint4*)&Wt[(size_t)(n0 + brow) * 384 + k0 + bcol];
            *(uint4*)&Bt[brow * LDK + bcol] = g;
        }
        __syncthreads();
        v16bf a  = load_frag_a(&As[wm * 16 * 32], 32);
        v16bf b0 = load_frag_b(&Bt[(wn * 32) * LDK]);
        v16bf b1 = load_frag_b(&Bt[(wn * 32 + 16) * LDK]);
        acc0 = __builtin_amdgcn_wmma_f32_16x16x32_bf16(false, a, false, b0, (short)0, acc0, false, false);
        acc1 = __builtin_amdgcn_wmma_f32_16x16x32_bf16(false, a, false, b1, (short)0, acc1, false, false);
        __syncthreads();
    }

    const int b   = m0 >> 12;
    const int l0a = (m0 & (LQ - 1)) + wm * 16 + 8 * half;     // lane's 8 l's contiguous
    const int n1  = n0 + wn * 32 + r;
    if (n0 < DIN) {                                            // -> xm_t (b,d,l)
        Acc8 u0, u1; u0.v = acc0; u1.v = acc1;
        float* p0 = &xm_t[((size_t)(b * DIN + n1)) * LQ + l0a];
        *(float4*)p0 = u0.f[0]; *(float4*)(p0 + 4) = u0.f[1];
        float* p1 = &xm_t[((size_t)(b * DIN + n1 + 16)) * LQ + l0a];
        *(float4*)p1 = u1.f[0]; *(float4*)(p1 + 4) = u1.f[1];
    } else {                                                   // -> z (b,l,d)
        const int zc = n1 - DIN;
#pragma unroll
        for (int j = 0; j < 8; ++j) {
            int l = l0a + j;
            z[((size_t)(b * LQ + l)) * DIN + zc]      = acc0[j];
            z[((size_t)(b * LQ + l)) * DIN + zc + 16] = acc1[j];
        }
    }
}

// -------------------------------------------------------------------------
// Depthwise 3x3 conv (pad 1) + bias + SiLU, layout (b, d, h, w) in/out
// -------------------------------------------------------------------------
__global__ __launch_bounds__(256) void k_conv(const float* __restrict__ xm_t,
                                              const float* __restrict__ cw,
                                              const float* __restrict__ cb,
                                              float* __restrict__ xc)
{
    int idx = blockIdx.x * 256 + threadIdx.x;
    if (idx >= BB * DIN * LQ) return;
    int wq = idx & 63;
    int h  = (idx >> 6) & 63;
    int d  = (idx >> 12) % DIN;
    int b  = idx / (DIN * LQ);
    const float* src = xm_t + ((size_t)(b * DIN + d)) * LQ;
    float acc = cb[d];
#pragma unroll
    for (int kh = 0; kh < 3; ++kh) {
        int hh = h + kh - 1;
        if (hh < 0 || hh > 63) continue;
#pragma unroll
        for (int kw = 0; kw < 3; ++kw) {
            int wn2 = wq + kw - 1;
            if (wn2 < 0 || wn2 > 63) continue;
            acc += src[hh * 64 + wn2] * cw[d * 9 + kh * 3 + kw];
        }
    }
    xc[idx] = acc / (1.f + __expf(-acc));              // SiLU
}

// -------------------------------------------------------------------------
// x_dbl = einsum('bkdl,kcd->bkcl').  A = x_proj_weight[k] (56x768 padded to 64),
// B = xs[b,k] gathered from xc via pos_map, staged column-major in LDS.
// -------------------------------------------------------------------------
__global__ __launch_bounds__(256) void k_xdbl(const float* __restrict__ xc,
                                              const float* __restrict__ xpw,
                                              float* __restrict__ xdbl)
{
    __shared__ __bf16 As[64 * 32];
    __shared__ __bf16 Bt[64 * LDK];
    const int l0 = blockIdx.x * 64;
    const int bk = blockIdx.y;
    const int b = bk >> 2, k = bk & 3;
    const int tid = threadIdx.x;
    const int lane = tid & 31, wv = tid >> 5;
    const int wm = wv >> 1, wn = wv & 1;
    const int half = lane >> 4, r = lane & 15;
    v8f acc0 = {}; v8f acc1 = {};

    const int arow = tid >> 3, acol = (tid & 7) * 4;

    for (int d0 = 0; d0 < DIN; d0 += 32) {
#pragma unroll
        for (int c = 0; c < 2; ++c) {                  // A: 56x32 padded to 64
            int row = arow + c * 32;
            float4 g = {0.f, 0.f, 0.f, 0.f};
            if (row < CPROJ)
                g = *(const float4*)&xpw[(size_t)(k * CPROJ + row) * DIN + d0 + acol];
            uint2 s; s.x = pack2(g.x, g.y); s.y = pack2(g.z, g.w);
            *(uint2*)&As[row * 32 + acol] = s;
        }
#pragma unroll
        for (int c = 0; c < 2; ++c) {                  // B: 32x64 gathered along l
            int e = tid + c * 256;
            int row = e >> 6, col = e & 63;            // row = d offset, col = l offset
            int p = pos_map(k, l0 + col);
            Bt[col * LDK + row] = (__bf16)xc[((size_t)(b * DIN + d0 + row)) * LQ + p];
        }
        __syncthreads();
        v16bf a  = load_frag_a(&As[wm * 16 * 32], 32);
        v16bf b0 = load_frag_b(&Bt[(wn * 32) * LDK]);
        v16bf b1 = load_frag_b(&Bt[(wn * 32 + 16) * LDK]);
        acc0 = __builtin_amdgcn_wmma_f32_16x16x32_bf16(false, a, false, b0, (short)0, acc0, false, false);
        acc1 = __builtin_amdgcn_wmma_f32_16x16x32_bf16(false, a, false, b1, (short)0, acc1, false, false);
        __syncthreads();
    }
#pragma unroll
    for (int j = 0; j < 8; ++j) {
        int c = wm * 16 + j + 8 * half;
        if (c >= CPROJ) continue;
        int n1 = l0 + wn * 32 + r;
        xdbl[((size_t)(bk * CPROJ + c)) * LQ + n1]      = acc0[j];
        xdbl[((size_t)(bk * CPROJ + c)) * LQ + n1 + 16] = acc1[j];
    }
}

__global__ void k_zero(float* __restrict__ p, int n)
{
    int i = blockIdx.x * 256 + threadIdx.x;
    if (i < n) p[i] = 0.f;
}

// -------------------------------------------------------------------------
// Selective scan: one thread per (b,k,d). dt-projection fused (wave-uniform
// loads). Direction merge via pos_map + coalesced f32 atomics, y is (b,l,d).
// -------------------------------------------------------------------------
__global__ __launch_bounds__(256) void k_scan(const float* __restrict__ xc,
                                              const float* __restrict__ xdbl,
                                              const float* __restrict__ dtw_g,
                                              const float* __restrict__ dtb_g,
                                              const float* __restrict__ Alogs,
                                              const float* __restrict__ Ds,
                                              float* __restrict__ y)
{
    int t = blockIdx.x * 256 + threadIdx.x;
    if (t >= BB * KDIR * DIN) return;
    const int d = t % DIN;
    const int k = (t / DIN) & 3;
    const int b = t / (KDIR * DIN);

    float dtw[RRK];
#pragma unroll
    for (int rr = 0; rr < RRK; ++rr) dtw[rr] = dtw_g[(size_t)(k * DIN + d) * RRK + rr];
    const float bias = dtb_g[k * DIN + d];
    float A[NST], h[NST];
#pragma unroll
    for (int n = 0; n < NST; ++n) {
        A[n] = -__expf(Alogs[(size_t)(k * DIN + d) * NST + n]);
        h[n] = 0.f;
    }
    const float Dp = Ds[k * DIN + d];
    const float* xdb  = xdbl + (size_t)(b * KDIR + k) * CPROJ * LQ;
    const float* xrow = xc   + (size_t)(b * DIN + d) * LQ;
    float* yb = y + (size_t)b * LQ * DIN;

    for (int l = 0; l < LQ; ++l) {
        const int p = pos_map(k, l);
        const float u = xrow[p];
        float dt = bias;
#pragma unroll
        for (int rr = 0; rr < RRK; ++rr) dt += xdb[(size_t)rr * LQ + l] * dtw[rr];
        const float delta = (dt > 20.f) ? dt : log1pf(__expf(dt));   // softplus
        const float du = delta * u;
        float acc = 0.f;
#pragma unroll
        for (int n = 0; n < NST; ++n) {
            const float bn = xdb[(size_t)(RRK + n) * LQ + l];
            const float cn = xdb[(size_t)(RRK + NST + n) * LQ + l];
            h[n] = h[n] * __expf(delta * A[n]) + du * bn;
            acc += h[n] * cn;
        }
        atomicAdd(&yb[(size_t)p * DIN + d], acc + Dp * u);
    }
}

// -------------------------------------------------------------------------
// LayerNorm over Din + SiLU(z) gate -> bf16 activations (b, l, d)
// -------------------------------------------------------------------------
__global__ __launch_bounds__(256) void k_ln(const float* __restrict__ y,
                                            const float* __restrict__ z,
                                            const float* __restrict__ g,
                                            const float* __restrict__ bta,
                                            __bf16* __restrict__ yg)
{
    __shared__ float red[256];
    const int row = blockIdx.x;
    const float* yr = y + (size_t)row * DIN;
    const float* zr = z + (size_t)row * DIN;
    __bf16* og = yg + (size_t)row * DIN;
    const int tid = threadIdx.x;

    float s = 0.f;
    for (int i = tid; i < DIN; i += 256) s += yr[i];
    red[tid] = s; __syncthreads();
    for (int st = 128; st > 0; st >>= 1) { if (tid < st) red[tid] += red[tid + st]; __syncthreads(); }
    const float mu = red[0] * (1.f / DIN);
    __syncthreads();

    float v = 0.f;
    for (int i = tid; i < DIN; i += 256) { float dd = yr[i] - mu; v += dd * dd; }
    red[tid] = v; __syncthreads();
    for (int st = 128; st > 0; st >>= 1) { if (tid < st) red[tid] += red[tid + st]; __syncthreads(); }
    const float inv = rsqrtf(red[0] * (1.f / DIN) + 1e-5f);

    for (int i = tid; i < DIN; i += 256) {
        const float yn = (yr[i] - mu) * inv * g[i] + bta[i];
        const float zv = zr[i];
        const float gate = zv / (1.f + __expf(-zv));
        og[i] = (__bf16)(yn * gate);
    }
}

// -------------------------------------------------------------------------
// GEMM out: out(32768x384) = yg(32768x768 bf16) @ W_out, B pre-transposed
// bf16 (384x768).
// -------------------------------------------------------------------------
__global__ __launch_bounds__(256) void k_gemm_out(const __bf16* __restrict__ yg,
                                                  const __bf16* __restrict__ Wt,
                                                  float* __restrict__ out)
{
    __shared__ __bf16 As[64 * 32];
    __shared__ __bf16 Bt[64 * LDK];
    const int m0 = blockIdx.x * 64;
    const int n0 = blockIdx.y * 64;
    const int tid = threadIdx.x;
    const int lane = tid & 31, wv = tid >> 5;
    const int wm = wv >> 1, wn = wv & 1;
    const int half = lane >> 4, r = lane & 15;
    v8f acc0 = {}; v8f acc1 = {};

    const int arow = tid >> 2, acol8 = (tid & 3) * 8;          // A: 1 uint4 / thread
    const int brow = tid >> 2, bcol8 = (tid & 3) * 8;

    for (int k0 = 0; k0 < DIN; k0 += 32) {
        if (k0 + 32 < DIN) {
            __builtin_prefetch(&yg[(size_t)(m0 + arow) * DIN + k0 + 32 + acol8], 0, 1);
            __builtin_prefetch(&Wt[(size_t)(n0 + brow) * DIN + k0 + 32 + bcol8], 0, 1);
        }
        {
            uint4 g = *(const uint4*)&yg[(size_t)(m0 + arow) * DIN + k0 + acol8];
            *(uint4*)&As[arow * 32 + acol8] = g;
        }
        {
            uint4 g = *(const uint4*)&Wt[(size_t)(n0 + brow) * DIN + k0 + bcol8];
            *(uint4*)&Bt[brow * LDK + bcol8] = g;
        }
        __syncthreads();
        v16bf a  = load_frag_a(&As[wm * 16 * 32], 32);
        v16bf b0 = load_frag_b(&Bt[(wn * 32) * LDK]);
        v16bf b1 = load_frag_b(&Bt[(wn * 32 + 16) * LDK]);
        acc0 = __builtin_amdgcn_wmma_f32_16x16x32_bf16(false, a, false, b0, (short)0, acc0, false, false);
        acc1 = __builtin_amdgcn_wmma_f32_16x16x32_bf16(false, a, false, b1, (short)0, acc1, false, false);
        __syncthreads();
    }
#pragma unroll
    for (int j = 0; j < 8; ++j) {
        int m = m0 + wm * 16 + j + 8 * half;
        int n1 = n0 + wn * 32 + r;
        out[(size_t)m * DD + n1]      = acc0[j];
        out[(size_t)m * DD + n1 + 16] = acc1[j];
    }
}

// -------------------------------------------------------------------------
extern "C" void kernel_launch(void* const* d_in, const int* in_sizes, int n_in,
                              void* d_out, int out_size, void* d_ws, size_t ws_size,
                              hipStream_t stream)
{
    (void)in_sizes; (void)n_in; (void)out_size; (void)ws_size;
    const float* x      = (const float*)d_in[0];
    const float* Win    = (const float*)d_in[1];
    const float* conv_w = (const float*)d_in[2];
    const float* conv_b = (const float*)d_in[3];
    const float* xpw    = (const float*)d_in[4];
    const float* dtw    = (const float*)d_in[5];
    const float* dtb    = (const float*)d_in[6];
    const float* Alogs  = (const float*)d_in[7];
    const float* Ds     = (const float*)d_in[8];
    const float* lng    = (const float*)d_in[9];
    const float* lnb    = (const float*)d_in[10];
    const float* Wout   = (const float*)d_in[11];
    float* out = (float*)d_out;

    const size_t SZ_BDL = (size_t)BB * DIN * LQ * sizeof(float);   // 100663296
    const size_t SZ_XDB = (size_t)BB * KDIR * CPROJ * LQ * sizeof(float);
    char* ws = (char*)d_ws;
    float*  xm_t  = (float*)(ws);                      // reused as y (b,l,d)
    float*  zbuf  = (float*)(ws + SZ_BDL);
    float*  xc    = (float*)(ws + 2 * SZ_BDL);         // reused as yg (bf16)
    float*  xdbl  = (float*)(ws + 3 * SZ_BDL);
    __bf16* Wtin  = (__bf16*)(ws + 3 * SZ_BDL + SZ_XDB);
    __bf16* Wtout = (__bf16*)(ws + 3 * SZ_BDL + SZ_XDB + (size_t)384 * 1536 * 2);
    float*  y     = xm_t;
    __bf16* yg    = (__bf16*)xc;

    const int NEL = BB * DIN * LQ;                     // 25165824

    k_transpose_cvt<<<(384 * 1536 + 255) / 256, 256, 0, stream>>>(Win, Wtin, 384, 1536);
    k_transpose_cvt<<<(768 * 384 + 255) / 256, 256, 0, stream>>>(Wout, Wtout, 768, 384);
    k_gemm_in<<<dim3((BB * LQ) / 64, (2 * DIN) / 64), 256, 0, stream>>>(x, Wtin, xm_t, zbuf);
    k_conv<<<NEL / 256, 256, 0, stream>>>(xm_t, conv_w, conv_b, xc);
    k_zero<<<NEL / 256, 256, 0, stream>>>(y, NEL);
    k_xdbl<<<dim3(LQ / 64, BB * KDIR), 256, 0, stream>>>(xc, xpw, xdbl);
    k_scan<<<(BB * KDIR * DIN) / 256, 256, 0, stream>>>(xc, xdbl, dtw, dtb, Alogs, Ds, y);
    k_ln<<<BB * LQ, 256, 0, stream>>>(y, zbuf, lng, lnb, yg);
    k_gemm_out<<<dim3((BB * LQ) / 64, DD / 64), 256, 0, stream>>>(yg, Wtout, out);
}
